// LogicityVisReasoningEngine_62732292325477
// MI455X (gfx1250) — matrix operand order repaired and checked by reading, not compiled, not run on hardware
//
#include <hip/hip_runtime.h>
#include <hip/hip_bf16.h>

// ---------------------------------------------------------------------------
// Problem constants (match reference)
// ---------------------------------------------------------------------------
static constexpr int cB   = 4;    // batch
static constexpr int cN   = 48;   // objects
static constexpr int FIMG = 2048;
static constexpr int CN_  = 8;    // node concepts
static constexpr int CE_  = 5;    // edge concepts (4 mlp + 1 priority)
static constexpr int CA_  = 4;
static constexpr int HID  = 8;
static constexpr float BBOX_MAX = 1024.0f;

typedef __attribute__((ext_vector_type(16))) _Float16 v16h;
typedef __attribute__((ext_vector_type(8)))  float    v8f;

__device__ __forceinline__ float sigmoidf_(float x) {
    return 1.0f / (1.0f + __expf(-x));
}

__device__ __forceinline__ void pack8_(v16h& d, int base, float4 lo, float4 hi) {
    d[base + 0] = (_Float16)lo.x;
    d[base + 1] = (_Float16)lo.y;
    d[base + 2] = (_Float16)lo.z;
    d[base + 3] = (_Float16)lo.w;
    d[base + 4] = (_Float16)hi.x;
    d[base + 5] = (_Float16)hi.y;
    d[base + 6] = (_Float16)hi.z;
    d[base + 7] = (_Float16)hi.w;
}

// ---------------------------------------------------------------------------
// WMMA GEMM:  C[M,N] = act( A[M,K] * B[K,N] + bias[N] )
// HARD REQUIREMENTS (all call sites satisfy): M % 16 == 0, K % 32 == 0,
// N % 16 == 0, A/B 16B-aligned.  No guards anywhere -> branchless inner loop:
// 8x global_load_b128 + 8x v_cvt_pk_f16_f32 + v_wmma_f32_16x16x32_f16.
// One wave (32 lanes) owns one 16x16 output tile; blockDim = (32, 4).
// act: 0 = none, 1 = relu, 2 = sigmoid
// ---------------------------------------------------------------------------
__global__ void gemm_f16w_kernel(const float* __restrict__ A,
                                 const float* __restrict__ Bm,
                                 const float* __restrict__ bias,
                                 float* __restrict__ C,
                                 int M, int K, int N, int act)
{
    const int lane = threadIdx.x;        // 0..31 (wave32)
    const int half = lane >> 4;          // 0 / 1
    const int l16  = lane & 15;
    const int tiles_n = N >> 4;
    const int tn = blockIdx.x * blockDim.y + threadIdx.y;
    const int tm = blockIdx.y;
    if (tn >= tiles_n) return;           // whole-wave uniform exit

    const int row = tm * 16 + l16;       // always < M
    // A fragment (16-bit A 16x32 layout): lane needs K offsets
    // k0 + half*8 + {0..7} and {16..23}  -> two contiguous float4 pairs.
    const float* ap = A + (size_t)row * K + half * 8;
    // B fragment: lane <-> column tn*16+l16, elements = K rows
    // k0 + half*8 + {0..7} and {16..23} (same order as A).
    const float* bp = Bm + (size_t)(half * 8) * N + tn * 16 + l16;
    const size_t bstep = (size_t)32 * N;

    v8f acc = {0.f, 0.f, 0.f, 0.f, 0.f, 0.f, 0.f, 0.f};

#pragma unroll 2
    for (int k0 = 0; k0 < K; k0 += 32) {
        float4 a0 = ((const float4*)ap)[0];   // +0..3
        float4 a1 = ((const float4*)ap)[1];   // +4..7
        float4 a2 = ((const float4*)ap)[4];   // +16..19
        float4 a3 = ((const float4*)ap)[5];   // +20..23
        float4 b0 = make_float4(bp[(size_t)0 * N], bp[(size_t)1 * N], bp[(size_t)2 * N], bp[(size_t)3 * N]);
        float4 b1 = make_float4(bp[(size_t)4 * N], bp[(size_t)5 * N], bp[(size_t)6 * N], bp[(size_t)7 * N]);
        float4 b2 = make_float4(bp[(size_t)16 * N], bp[(size_t)17 * N], bp[(size_t)18 * N], bp[(size_t)19 * N]);
        float4 b3 = make_float4(bp[(size_t)20 * N], bp[(size_t)21 * N], bp[(size_t)22 * N], bp[(size_t)23 * N]);
        ap += 32;
        bp += bstep;

        v16h a, b;
        pack8_(a, 0, a0, a1);
        pack8_(a, 8, a2, a3);
        pack8_(b, 0, b0, b1);
        pack8_(b, 8, b2, b3);

        acc = __builtin_amdgcn_wmma_f32_16x16x32_f16(
                  /*neg_a=*/false, a, /*neg_b=*/false, b,
                  /*c_mod=*/(short)0, acc, /*reuse_a=*/false, /*reuse_b=*/false);
    }

    // C/D layout: element e -> M = tm*16 + e + 8*half, lane -> N = tn*16 + l16
    const int col = tn * 16 + l16;
    const float bc = bias[col];
    float* Cp = C + (size_t)(tm * 16 + 8 * half) * N + col;
#pragma unroll
    for (int e = 0; e < 8; ++e) {
        float v = acc[e] + bc;
        if (act == 1)      v = fmaxf(v, 0.f);
        else if (act == 2) v = sigmoidf_(v);
        Cp[(size_t)e * N] = v;
    }
}

// ---------------------------------------------------------------------------
// Zero-pad the (16,256) edge weight to (32,256) so edge GEMM 1 has K%32==0.
// ---------------------------------------------------------------------------
__global__ void pad_w0_kernel(const float* __restrict__ w,
                              float* __restrict__ wp)
{
    int t = blockIdx.x * blockDim.x + threadIdx.x;
    if (t >= 32 * 256) return;
    int r = t / 256;
    wp[t] = (r < 16) ? w[t] : 0.f;
}

// ---------------------------------------------------------------------------
// Zero-pad columns of (K,Nsrc) to (K,16)  (used for N=8 / N=4 weights+biases).
// ---------------------------------------------------------------------------
__global__ void pad_cols16_kernel(const float* __restrict__ src,
                                  float* __restrict__ dst, int K, int Nsrc)
{
    int t = blockIdx.x * blockDim.x + threadIdx.x;
    if (t >= K * 16) return;
    int r = t / 16, c = t % 16;
    dst[t] = (c < Nsrc) ? src[r * Nsrc + c] : 0.f;
}

// ---------------------------------------------------------------------------
// Compact columns: (M,16) -> (M,Ndst), dropping padded columns.
// ---------------------------------------------------------------------------
__global__ void compact_cols_kernel(const float* __restrict__ src,
                                    float* __restrict__ dst, int M, int Ndst)
{
    int t = blockIdx.x * blockDim.x + threadIdx.x;
    if (t >= M * Ndst) return;
    int r = t / Ndst, c = t % Ndst;
    dst[t] = src[(size_t)r * 16 + c];
}

// ---------------------------------------------------------------------------
// Edge input features, K padded to 32:
// full[b,i,j,0:16] = [attr(b,i), attr(b,j)], [16:32] = 0
// attr(b,n) = [bbox(b,n,:)/1024, dir(b,n,:)]
// ---------------------------------------------------------------------------
__global__ void build_edge_in_kernel(const float* __restrict__ bbox,
                                     const float* __restrict__ dir,
                                     float* __restrict__ out)
{
    int idx = blockIdx.x * blockDim.x + threadIdx.x;
    if (idx >= cB * cN * cN) return;
    int j = idx % cN;
    int t = idx / cN;
    int i = t % cN;
    int b = t / cN;
    float* dst = out + (size_t)idx * 32;
#pragma unroll
    for (int c = 0; c < 4; ++c) {
        dst[c]      = bbox[((size_t)b * cN + i) * 4 + c] * (1.0f / BBOX_MAX);
        dst[4 + c]  = dir [((size_t)b * cN + i) * 4 + c];
        dst[8 + c]  = bbox[((size_t)b * cN + j) * 4 + c] * (1.0f / BBOX_MAX);
        dst[12 + c] = dir [((size_t)b * cN + j) * 4 + c];
    }
#pragma unroll
    for (int c = 16; c < 32; ++c) dst[c] = 0.f;
}

// ---------------------------------------------------------------------------
// edge_full[b,i,j,0:5] = diag? 0 : [edge_mlp_sigmoid(4), pri_i > pri_j]
// ---------------------------------------------------------------------------
__global__ void build_edge_full_kernel(const float* __restrict__ e3,
                                       const float* __restrict__ pri,
                                       float* __restrict__ edge_full)
{
    int idx = blockIdx.x * blockDim.x + threadIdx.x;
    if (idx >= cB * cN * cN) return;
    int j = idx % cN;
    int t = idx / cN;
    int i = t % cN;
    int b = t / cN;
    float* dst = edge_full + (size_t)idx * CE_;
    if (i == j) {
#pragma unroll
        for (int c = 0; c < CE_; ++c) dst[c] = 0.f;
    } else {
#pragma unroll
        for (int c = 0; c < CE_ - 1; ++c) dst[c] = e3[(size_t)idx * (CE_ - 1) + c];
        dst[CE_ - 1] = (pri[b * cN + i] > pri[b * cN + j]) ? 1.f : 0.f;
    }
}

// ---------------------------------------------------------------------------
// red3[b,i,j, 2c/2c+1] = max_k / min_k of f3[b,i,j,k,c] with diag (k==j)
// replaced by 0 (exists) / 1 (forall).  f3 channels = HID.
// ---------------------------------------------------------------------------
__global__ void reduce3_kernel(const float* __restrict__ f3,
                               float* __restrict__ red3)
{
    int idx = blockIdx.x * blockDim.x + threadIdx.x;
    if (idx >= cB * cN * cN) return;
    int j = idx % cN;
    float mx[HID], mn[HID];
#pragma unroll
    for (int c = 0; c < HID; ++c) { mx[c] = -1e30f; mn[c] = 1e30f; }
    const float* base = f3 + (size_t)idx * cN * HID;
    for (int k = 0; k < cN; ++k) {
        const float* row = base + (size_t)k * HID;
        bool diag = (k == j);
#pragma unroll
        for (int c = 0; c < HID; ++c) {
            float v = row[c];
            float e = diag ? 0.f : v;
            float f = diag ? 1.f : v;
            mx[c] = fmaxf(mx[c], e);
            mn[c] = fminf(mn[c], f);
        }
    }
    float* dst = red3 + (size_t)idx * (2 * HID);
#pragma unroll
    for (int c = 0; c < HID; ++c) { dst[2 * c] = mx[c]; dst[2 * c + 1] = mn[c]; }
}

// ---------------------------------------------------------------------------
// Arity-0 logic layer: out0[b,:] = sigmoid([f0(C0) , reduce(f1)(2*C1)] @ W + b)
// ---------------------------------------------------------------------------
__global__ void nlm_arity0_kernel(const float* __restrict__ f0, int C0,
                                  const float* __restrict__ f1, int C1,
                                  const float* __restrict__ W,
                                  const float* __restrict__ bias,
                                  float* __restrict__ out0)
{
    int t = threadIdx.x;
    if (t >= cB * HID) return;
    int b = t / HID, o = t % HID;
    float acc = bias[o];
    int r = 0;
    for (int c = 0; c < C0; ++c) { acc += f0[b * HID + c] * W[r * HID + o]; ++r; }
    for (int c = 0; c < C1; ++c) {
        float mx = -1e30f, mn = 1e30f;
        for (int i = 0; i < cN; ++i) {
            float v = f1[((size_t)b * cN + i) * C1 + c];
            mx = fmaxf(mx, v); mn = fminf(mn, v);
        }
        acc += mx * W[r * HID + o]; ++r;
        acc += mn * W[r * HID + o]; ++r;
    }
    out0[t] = sigmoidf_(acc);
}

// ---------------------------------------------------------------------------
// Arity-1: input per (b,i) = [f0(C0), f1(C1), reduce(f2)(2*C2)]
// ---------------------------------------------------------------------------
__global__ void nlm_arity1_kernel(const float* __restrict__ f0, int C0,
                                  const float* __restrict__ f1, int C1,
                                  const float* __restrict__ f2, int C2,
                                  const float* __restrict__ W,
                                  const float* __restrict__ bias,
                                  float* __restrict__ out1)
{
    int idx = blockIdx.x * blockDim.x + threadIdx.x;
    if (idx >= cB * cN) return;
    int b = idx / cN, i = idx % cN;
    float acc[HID];
#pragma unroll
    for (int o = 0; o < HID; ++o) acc[o] = bias[o];
    int r = 0;
    for (int c = 0; c < C0; ++c) {
        float v = f0[b * HID + c];
#pragma unroll
        for (int o = 0; o < HID; ++o) acc[o] += v * W[r * HID + o];
        ++r;
    }
    for (int c = 0; c < C1; ++c) {
        float v = f1[(size_t)idx * C1 + c];
#pragma unroll
        for (int o = 0; o < HID; ++o) acc[o] += v * W[r * HID + o];
        ++r;
    }
    for (int c = 0; c < C2; ++c) {
        float mx = -1e30f, mn = 1e30f;
        for (int j = 0; j < cN; ++j) {
            if (j == i) { mx = fmaxf(mx, 0.f); mn = fminf(mn, 1.f); }
            else {
                float v = f2[(((size_t)b * cN + i) * cN + j) * C2 + c];
                mx = fmaxf(mx, v); mn = fminf(mn, v);
            }
        }
#pragma unroll
        for (int o = 0; o < HID; ++o) acc[o] += mx * W[r * HID + o];
        ++r;
#pragma unroll
        for (int o = 0; o < HID; ++o) acc[o] += mn * W[r * HID + o];
        ++r;
    }
#pragma unroll
    for (int o = 0; o < HID; ++o) out1[(size_t)idx * HID + o] = sigmoidf_(acc[o]);
}

// ---------------------------------------------------------------------------
// Arity-2: per (b,i,j), two permutations (i,j),(j,i) of
// cat = [expand(f1)(C1), f2(C2), reduce(f3)(2*HID if red3)]
// ---------------------------------------------------------------------------
__global__ void nlm_arity2_kernel(const float* __restrict__ f1, int C1,
                                  const float* __restrict__ f2, int C2,
                                  const float* __restrict__ red3, // may be null
                                  const float* __restrict__ W,
                                  const float* __restrict__ bias,
                                  float* __restrict__ out2)
{
    __shared__ float Ws[64 * HID];
    __shared__ float bs[HID];
    const int Kdim = 2 * (C1 + C2 + (red3 ? 2 * HID : 0));
    for (int t = threadIdx.x; t < Kdim * HID; t += blockDim.x) Ws[t] = W[t];
    if (threadIdx.x < HID) bs[threadIdx.x] = bias[threadIdx.x];
    __syncthreads();

    int idx = blockIdx.x * blockDim.x + threadIdx.x;
    if (idx >= cB * cN * cN) return;
    int j = idx % cN;
    int t2 = idx / cN;
    int i = t2 % cN;
    int b = t2 / cN;

    float acc[HID];
#pragma unroll
    for (int o = 0; o < HID; ++o) acc[o] = bs[o];

    int X[2] = { i, j };
    int Y[2] = { j, i };
    int r = 0;
#pragma unroll
    for (int p = 0; p < 2; ++p) {
        const float* f1row = f1 + ((size_t)b * cN + X[p]) * C1;
        for (int c = 0; c < C1; ++c) {
            float v = f1row[c];
#pragma unroll
            for (int o = 0; o < HID; ++o) acc[o] += v * Ws[r * HID + o];
            ++r;
        }
        const float* f2row = f2 + (((size_t)b * cN + X[p]) * cN + Y[p]) * C2;
        for (int c = 0; c < C2; ++c) {
            float v = f2row[c];
#pragma unroll
            for (int o = 0; o < HID; ++o) acc[o] += v * Ws[r * HID + o];
            ++r;
        }
        if (red3) {
            const float* r3 = red3 + (((size_t)b * cN + X[p]) * cN + Y[p]) * (2 * HID);
            for (int c = 0; c < 2 * HID; ++c) {
                float v = r3[c];
#pragma unroll
                for (int o = 0; o < HID; ++o) acc[o] += v * Ws[r * HID + o];
                ++r;
            }
        }
    }
#pragma unroll
    for (int o = 0; o < HID; ++o) out2[(size_t)idx * HID + o] = sigmoidf_(acc[o]);
}

// ---------------------------------------------------------------------------
// Arity-3: per (b,i,j,k), six permutations of cat = [expand(f2)(C2), f3(C3)]
// perm index maps: (i,j,k),(i,k,j),(j,i,k),(k,i,j),(j,k,i),(k,j,i)
// ---------------------------------------------------------------------------
__global__ void nlm_arity3_kernel(const float* __restrict__ f2, int C2,
                                  const float* __restrict__ f3, int C3,
                                  const float* __restrict__ W,
                                  const float* __restrict__ bias,
                                  float* __restrict__ out3)
{
    __shared__ float Ws[96 * HID];
    __shared__ float bs[HID];
    const int Kdim = 6 * (C2 + C3);
    for (int t = threadIdx.x; t < Kdim * HID; t += blockDim.x) Ws[t] = W[t];
    if (threadIdx.x < HID) bs[threadIdx.x] = bias[threadIdx.x];
    __syncthreads();

    int idx = blockIdx.x * blockDim.x + threadIdx.x;
    if (idx >= cB * cN * cN * cN) return;
    int k = idx % cN;
    int t3 = idx / cN;
    int j = t3 % cN;
    t3 /= cN;
    int i = t3 % cN;
    int b = t3 / cN;

    int X[6] = { i, i, j, k, j, k };
    int Y[6] = { j, k, i, i, k, j };
    int Z[6] = { k, j, k, j, i, i };

    float acc[HID];
#pragma unroll
    for (int o = 0; o < HID; ++o) acc[o] = bs[o];

    int r = 0;
#pragma unroll
    for (int p = 0; p < 6; ++p) {
        const float* f2row = f2 + (((size_t)b * cN + X[p]) * cN + Y[p]) * C2;
        for (int c = 0; c < C2; ++c) {
            float v = f2row[c];
#pragma unroll
            for (int o = 0; o < HID; ++o) acc[o] += v * Ws[r * HID + o];
            ++r;
        }
        if (C3 > 0) {
            const float* f3row =
                f3 + ((((size_t)b * cN + X[p]) * cN + Y[p]) * cN + Z[p]) * C3;
            for (int c = 0; c < C3; ++c) {
                float v = f3row[c];
#pragma unroll
                for (int o = 0; o < HID; ++o) acc[o] += v * Ws[r * HID + o];
                ++r;
            }
        }
    }
#pragma unroll
    for (int o = 0; o < HID; ++o) out3[(size_t)idx * HID + o] = sigmoidf_(acc[o]);
}

// ---------------------------------------------------------------------------
// Final head: pred[i,o] = f1_final[0,i,:] @ pred_W + pred_b (no activation)
// ---------------------------------------------------------------------------
__global__ void pred_kernel(const float* __restrict__ f1fin,
                            const float* __restrict__ pW,
                            const float* __restrict__ pb,
                            float* __restrict__ out)
{
    int t = threadIdx.x;
    if (t >= cN * CA_) return;
    int i = t / CA_, o = t % CA_;
    float acc = pb[o];
#pragma unroll
    for (int c = 0; c < HID; ++c)
        acc += f1fin[(size_t)i * HID + c] * pW[c * CA_ + o];
    out[t] = acc;
}

// ---------------------------------------------------------------------------
// Output assembly: node_concepts[0] and edge_compact[0]
// d_out layout: [pred 48*4][node 48*8][edge_compact 48*47*5]
// ---------------------------------------------------------------------------
__global__ void copy_out_kernel(const float* __restrict__ node_c,
                                const float* __restrict__ edge_full,
                                float* __restrict__ out)
{
    const int NODE_CNT = cN * CN_;                 // 384
    const int EDGE_CNT = cN * (cN - 1) * CE_;      // 11280
    int t = blockIdx.x * blockDim.x + threadIdx.x;
    if (t < NODE_CNT) {
        out[cN * CA_ + t] = node_c[t];             // batch 0 rows are first
    } else if (t < NODE_CNT + EDGE_CNT) {
        int e = t - NODE_CNT;
        int c  = e % CE_;
        int tt = e / CE_;
        int kk = tt % (cN - 1);
        int i  = tt / (cN - 1);
        int j  = kk + (kk >= i ? 1 : 0);
        out[cN * CA_ + NODE_CNT + e] =
            edge_full[(((size_t)0 * cN + i) * cN + j) * CE_ + c];
    }
}

// ---------------------------------------------------------------------------
// Host side
// ---------------------------------------------------------------------------
static void launch_gemm(const float* A, const float* Bm, const float* bias,
                        float* C, int M, int K, int N, int act, hipStream_t s)
{
    int tiles_m = M / 16;
    int tiles_n = N / 16;
    dim3 block(32, 4);
    dim3 grid((tiles_n + 3) / 4, tiles_m);
    gemm_f16w_kernel<<<grid, block, 0, s>>>(A, Bm, bias, C, M, K, N, act);
}

extern "C" void kernel_launch(void* const* d_in, const int* in_sizes, int n_in,
                              void* d_out, int out_size, void* d_ws, size_t ws_size,
                              hipStream_t stream)
{
    (void)in_sizes; (void)n_in; (void)out_size; (void)ws_size;

    const float* roi  = (const float*)d_in[0];   // (4,48,2048)
    const float* bbox = (const float*)d_in[1];   // (4,48,4)
    const float* dirs = (const float*)d_in[2];   // (4,48,4)
    const float* pri  = (const float*)d_in[3];   // (4,48)
    const float* nW0 = (const float*)d_in[4];  const float* nb0 = (const float*)d_in[5];
    const float* nW1 = (const float*)d_in[6];  const float* nb1 = (const float*)d_in[7];
    const float* nW2 = (const float*)d_in[8];  const float* nb2 = (const float*)d_in[9];
    const float* eW0 = (const float*)d_in[10]; const float* eb0 = (const float*)d_in[11];
    const float* eW1 = (const float*)d_in[12]; const float* eb1 = (const float*)d_in[13];
    const float* eW2 = (const float*)d_in[14]; const float* eb2 = (const float*)d_in[15];
    auto nlmW = [&](int l, int i) { return (const float*)d_in[16 + (l * 4 + i) * 2]; };
    auto nlmB = [&](int l, int i) { return (const float*)d_in[17 + (l * 4 + i) * 2]; };
    const float* predW = (const float*)d_in[48];
    const float* predB = (const float*)d_in[49];

    // ---- workspace carve-out (floats, rounded to 64B for float4 alignment) ----
    float* p = (float*)d_ws;
    auto alloc = [&](size_t n) {
        n = (n + 15) & ~(size_t)15;
        float* q = p; p += n; return q;
    };
    const int M_NODE = cB * cN;            // 192
    const int M_EDGE = cB * cN * cN;       // 9216
    const size_t N3   = (size_t)cB * cN * cN * cN; // 442368

    float* h1        = alloc((size_t)M_NODE * 512);
    float* h2        = alloc((size_t)M_NODE * 256);
    float* node_c16  = alloc((size_t)M_NODE * 16);   // padded-N output
    float* node_c    = alloc((size_t)M_NODE * CN_);
    float* edge_in   = alloc((size_t)M_EDGE * 32);   // K padded 16 -> 32
    float* w0p       = alloc((size_t)32 * 256);      // zero-padded eW0 (K pad)
    float* nW2p      = alloc((size_t)256 * 16);      // zero-padded nW2 (N pad)
    float* nb2p      = alloc((size_t)16);
    float* eW2p      = alloc((size_t)64 * 16);       // zero-padded eW2 (N pad)
    float* eb2p      = alloc((size_t)16);
    float* eh1       = alloc((size_t)M_EDGE * 256);
    float* eh2       = alloc((size_t)M_EDGE * 64);
    float* e3_16     = alloc((size_t)M_EDGE * 16);   // padded-N output
    float* e3        = alloc((size_t)M_EDGE * (CE_ - 1));
    float* edge_full = alloc((size_t)M_EDGE * CE_);
    float* f0a = alloc(cB * HID);             float* f0b = alloc(cB * HID);
    float* f1a = alloc((size_t)M_NODE * HID); float* f1b = alloc((size_t)M_NODE * HID);
    float* f2a = alloc((size_t)M_EDGE * HID); float* f2b = alloc((size_t)M_EDGE * HID);
    float* f3a = alloc(N3 * HID);             float* f3b = alloc(N3 * HID);
    float* red3 = alloc((size_t)M_EDGE * 2 * HID);

    const int T256 = 256;
    const int gEdge = (M_EDGE + T256 - 1) / T256;          // 36
    const int gN3   = (int)((N3 + T256 - 1) / T256);       // 1728
    auto g1d = [&](int n) { return (n + T256 - 1) / T256; };

    // ---- weight padding (tiny, once per call) ----
    pad_w0_kernel<<<g1d(32 * 256), T256, 0, stream>>>(eW0, w0p);
    pad_cols16_kernel<<<g1d(256 * 16), T256, 0, stream>>>(nW2, nW2p, 256, CN_);
    pad_cols16_kernel<<<1, 16, 0, stream>>>(nb2, nb2p, 1, CN_);
    pad_cols16_kernel<<<g1d(64 * 16), T256, 0, stream>>>(eW2, eW2p, 64, CE_ - 1);
    pad_cols16_kernel<<<1, 16, 0, stream>>>(eb2, eb2p, 1, CE_ - 1);

    // ---- Node MLP (WMMA GEMMs) ----
    launch_gemm(roi, nW0, nb0, h1,       M_NODE, FIMG, 512, 1, stream);
    launch_gemm(h1,  nW1, nb1, h2,       M_NODE, 512,  256, 1, stream);
    launch_gemm(h2,  nW2p, nb2p, node_c16, M_NODE, 256, 16, 2, stream);
    compact_cols_kernel<<<g1d(M_NODE * CN_), T256, 0, stream>>>(
        node_c16, node_c, M_NODE, CN_);

    // ---- Edge MLP (WMMA GEMMs, K of first layer padded to 32) ----
    build_edge_in_kernel<<<gEdge, T256, 0, stream>>>(bbox, dirs, edge_in);
    launch_gemm(edge_in, w0p, eb0, eh1,  M_EDGE, 32,  256, 1, stream);
    launch_gemm(eh1,     eW1, eb1, eh2,  M_EDGE, 256, 64,  1, stream);
    launch_gemm(eh2,     eW2p, eb2p, e3_16, M_EDGE, 64, 16, 2, stream);
    compact_cols_kernel<<<g1d(M_EDGE * (CE_ - 1)), T256, 0, stream>>>(
        e3_16, e3, M_EDGE, CE_ - 1);
    build_edge_full_kernel<<<gEdge, T256, 0, stream>>>(e3, pri, edge_full);

    // ---- NLM layer 0: dims [0, CN=8, CE=5, 0] ----
    nlm_arity0_kernel<<<1, 32, 0, stream>>>(nullptr, 0, node_c, CN_,
                                            nlmW(0, 0), nlmB(0, 0), f0a);
    nlm_arity1_kernel<<<1, T256, 0, stream>>>(nullptr, 0, node_c, CN_,
                                              edge_full, CE_,
                                              nlmW(0, 1), nlmB(0, 1), f1a);
    nlm_arity2_kernel<<<gEdge, T256, 0, stream>>>(node_c, CN_, edge_full, CE_,
                                                  nullptr,
                                                  nlmW(0, 2), nlmB(0, 2), f2a);
    nlm_arity3_kernel<<<gN3, T256, 0, stream>>>(edge_full, CE_, nullptr, 0,
                                                nlmW(0, 3), nlmB(0, 3), f3a);

    // ---- NLM layers 1..3: dims [8,8,8,8], ping-pong a <-> b ----
    for (int l = 1; l < 4; ++l) {
        float* i0 = (l % 2 == 1) ? f0a : f0b;  float* o0 = (l % 2 == 1) ? f0b : f0a;
        float* i1 = (l % 2 == 1) ? f1a : f1b;  float* o1 = (l % 2 == 1) ? f1b : f1a;
        float* i2 = (l % 2 == 1) ? f2a : f2b;  float* o2 = (l % 2 == 1) ? f2b : f2a;
        float* i3 = (l % 2 == 1) ? f3a : f3b;  float* o3 = (l % 2 == 1) ? f3b : f3a;

        reduce3_kernel<<<gEdge, T256, 0, stream>>>(i3, red3);
        nlm_arity0_kernel<<<1, 32, 0, stream>>>(i0, HID, i1, HID,
                                                nlmW(l, 0), nlmB(l, 0), o0);
        nlm_arity1_kernel<<<1, T256, 0, stream>>>(i0, HID, i1, HID, i2, HID,
                                                  nlmW(l, 1), nlmB(l, 1), o1);
        nlm_arity2_kernel<<<gEdge, T256, 0, stream>>>(i1, HID, i2, HID, red3,
                                                      nlmW(l, 2), nlmB(l, 2), o2);
        nlm_arity3_kernel<<<gN3, T256, 0, stream>>>(i2, HID, i3, HID,
                                                    nlmW(l, 3), nlmB(l, 3), o3);
    }

    // final f1 lives in f1b (layer0->a, 1->b, 2->a, 3->b)
    pred_kernel<<<1, cN * CA_, 0, stream>>>(f1b, predW, predB, (float*)d_out);

    const int copy_total = cN * CN_ + cN * (cN - 1) * CE_;
    copy_out_kernel<<<(copy_total + T256 - 1) / T256, T256, 0, stream>>>(
        node_c, edge_full, (float*)d_out);
}
